// MixtureOfExpertsBlock_21680994910411
// MI455X (gfx1250) — compile-verified, matrix operand order checked by reading, not run on hardware
//
#include <hip/hip_runtime.h>
#include <hip/hip_bf16.h>
#include <stdint.h>

// ---------------- problem constants ----------------
#define BB 2
#define SS 2048
#define DD 1024
#define EE 8
#define HH 4096
#define NTOK (BB*SS)          // 4096 tokens
#define MT   32               // tokens per block tile (2 x 16-row WMMA M-tiles)
#define NTILE (NTOK/MT)       // 128 token tiles
#define NTHR 1024             // 32 waves (full WGP) -> low per-wave register pressure

typedef __bf16 bf16_t;
typedef __attribute__((ext_vector_type(16))) __bf16 v16bf;
typedef __attribute__((ext_vector_type(8)))  float  v8f;

// fragment swizzle helpers (bf16 16x16x32 WMMA, wave32):
//   lane = (m_or_n & 15) + ((kloc>>3)&1)*16
//   slot = ((kloc>>4)&1)*8 + (kloc&7)        (slot = bf16 index within v16bf)
__device__ __forceinline__ int frag_lane(int mn, int kloc) {
    return (mn & 15) + (((kloc >> 3) & 1) << 4);
}
__device__ __forceinline__ int frag_slot(int kloc) {
    return (((kloc >> 4) & 1) << 3) + (kloc & 7);
}

// ---------------- weight conversion: W1 [E][D][H] f32 -> B-fragments bf16 ----------------
// layout: W1ws[((e*256 + npanel)*32 + kstep)*512 + lane*16 + slot], npanel=h/16, kstep=d/32
__global__ __launch_bounds__(256) void conv_w1(const float* __restrict__ W1,
                                               bf16_t* __restrict__ W1ws) {
    size_t i = (size_t)blockIdx.x * 256 + threadIdx.x;   // over E*D*H
    int h = (int)(i % HH);
    size_t t = i / HH;
    int d = (int)(t % DD);
    int e = (int)(t / DD);
    int np = h >> 4, n = h & 15;
    int ks = d >> 5, kloc = d & 31;
    size_t dst = (((size_t)e * 256 + np) * 32 + ks) * 512
               + frag_lane(n, kloc) * 16 + frag_slot(kloc);
    W1ws[dst] = (bf16_t)W1[i];
}

// W2 [E][H][D] f32 -> B-fragments bf16
// layout: W2ws[((e*64 + npanel)*128 + kstep)*512 + lane*16 + slot], npanel=d/16, kstep=h/32
__global__ __launch_bounds__(256) void conv_w2(const float* __restrict__ W2,
                                               bf16_t* __restrict__ W2ws) {
    size_t i = (size_t)blockIdx.x * 256 + threadIdx.x;   // over E*H*D
    int d = (int)(i % DD);
    size_t t = i / DD;
    int h = (int)(t % HH);
    int e = (int)(t / HH);
    int np = d >> 4, n = d & 15;
    int ks = h >> 5, kloc = h & 31;
    size_t dst = (((size_t)e * 64 + np) * 128 + ks) * 512
               + frag_lane(n, kloc) * 16 + frag_slot(kloc);
    W2ws[dst] = (bf16_t)W2[i];
}

// ---------------- gating: top-2 softmax weights, already divided by E ----------------
__global__ __launch_bounds__(64) void gate_kernel(const float* __restrict__ x,
                                                  const float* __restrict__ Wg,
                                                  float* __restrict__ wmask) {
    int token = blockIdx.x * 2 + (threadIdx.x >> 5);
    int lane  = threadIdx.x & 31;
    const float* xr = x + (size_t)token * DD;
    float g[EE];
#pragma unroll
    for (int e = 0; e < EE; ++e) g[e] = 0.f;
    for (int d = lane; d < DD; d += 32) {
        float xv = xr[d];
#pragma unroll
        for (int e = 0; e < EE; ++e) g[e] += xv * Wg[d * EE + e];
    }
#pragma unroll
    for (int off = 16; off > 0; off >>= 1) {
#pragma unroll
        for (int e = 0; e < EE; ++e) g[e] += __shfl_xor(g[e], off, 32);
    }
    if (lane == 0) {
        float m1 = -3.4e38f, m2 = -3.4e38f;
#pragma unroll
        for (int e = 0; e < EE; ++e) {
            float v = g[e];
            if (v > m1) { m2 = m1; m1 = v; } else if (v > m2) { m2 = v; }
        }
        float p[EE], sum = 0.f;
#pragma unroll
        for (int e = 0; e < EE; ++e) {
            float v = (g[e] >= m2) ? __expf(g[e] - m1) : 0.f;
            p[e] = v; sum += v;
        }
        float inv = 1.f / (sum * (float)EE);   // fold mean-over-experts here
#pragma unroll
        for (int e = 0; e < EE; ++e) wmask[token * EE + e] = p[e] * inv;
    }
}

// ---------------- output init: out = sum_e w[tok][e] * b2[e][d] ----------------
__global__ __launch_bounds__(256) void out_bias_init(const float* __restrict__ b2,
                                                     const float* __restrict__ wmask,
                                                     float* __restrict__ out) {
    size_t i = (size_t)blockIdx.x * 256 + threadIdx.x;   // over NTOK*DD
    int d = (int)(i % DD);
    int token = (int)(i / DD);
    float acc = 0.f;
#pragma unroll
    for (int e = 0; e < EE; ++e)
        acc += wmask[token * EE + e] * b2[e * DD + d];
    out[i] = acc;
}

// ---------------- fused MoE FFN: (32-token tile, expert) per block, 32 waves ----------------
// dynamic LDS layout:
//   xA : 32 ksteps x 2 mtiles x 512 bf16   (64 KB)  A fragments of x tile
//   hA : 16 ksteps x 2 mtiles x 512 bf16   (32 KB)  A fragments of h chunk
//   wm : 32*8 float                        ( 1 KB)
#define SMEM_XA_ELEMS (32*2*512)
#define SMEM_HA_ELEMS (16*2*512)
#define SMEM_BYTES (SMEM_XA_ELEMS*2 + SMEM_HA_ELEMS*2 + MT*EE*4)

__global__ __launch_bounds__(NTHR) void moe_ffn(const float* __restrict__ x,
                                                const float* __restrict__ b1,
                                                const bf16_t* __restrict__ W1ws,
                                                const bf16_t* __restrict__ W2ws,
                                                const float* __restrict__ wmask,
                                                float* __restrict__ out) {
    const int tile = blockIdx.x;        // 0..127
    const int e    = blockIdx.y;        // 0..7
    const int tok0 = tile * MT;
    const int tid  = threadIdx.x;
    const int lane = tid & 31;
    const int wave = tid >> 5;          // 0..31

    extern __shared__ __align__(64) char smem_raw[];
    bf16_t* xA = (bf16_t*)smem_raw;
    bf16_t* hA = xA + SMEM_XA_ELEMS;
    float*  wm = (float*)(hA + SMEM_HA_ELEMS);
    __shared__ int s_active;

    if (tid < MT * EE) wm[tid] = wmask[tok0 * EE + tid];
    __syncthreads();
    if (tid == 0) {
        int a = 0;
        for (int i = 0; i < MT; ++i) a |= (wm[i * EE + e] != 0.f) ? 1 : 0;
        s_active = a;
    }
    __syncthreads();
    if (!s_active) return;   // this expert's contribution to this tile is exactly zero

    // load + bf16-convert + swizzle x tile into A fragments
    for (int i = tid; i < MT * DD; i += NTHR) {
        int m = i >> 10;                 // 0..31
        int d = i & (DD - 1);
        int mt = m >> 4, mr = m & 15;
        float f = x[(size_t)(tok0 + m) * DD + d];
        int ks = d >> 5, kloc = d & 31;
        xA[(ks * 2 + mt) * 512 + frag_lane(mr, kloc) * 16 + frag_slot(kloc)] = (bf16_t)f;
    }
    __syncthreads();

    v8f acc[2][2];                       // 2 D-panels x 2 M-tiles = 32 VGPRs
#pragma unroll
    for (int j = 0; j < 2; ++j) { acc[j][0] = {}; acc[j][1] = {}; }

    const int n     = lane & 15;
    const int mbase = (lane >> 4) * 8;

    const bf16_t* W1e = W1ws + (size_t)e * (256 * 32 * 512);
    const bf16_t* W2e = W2ws + (size_t)e * (64 * 128 * 512);

    for (int hc = 0; hc < 8; ++hc) {     // hidden chunks of 512
        __syncthreads();                  // hA safe to overwrite
        // ---- GEMM1: h_chunk[32,512] = relu(x @ W1 + b1) * gate_w ----
        {
            const int np = hc * 32 + wave;            // H panel 0..255, 1 per wave
            v8f c0 = {}, c1 = {};
            const bf16_t* bp = W1e + ((size_t)np * 32) * 512 + lane * 16;
            const bf16_t* ap = xA + lane * 16;
            for (int kk = 0; kk < 32; ++kk) {
                v16bf b  = *(const v16bf*)(bp + (size_t)kk * 512);
                v16bf a0 = *(const v16bf*)(ap + (kk * 2 + 0) * 512);
                v16bf a1 = *(const v16bf*)(ap + (kk * 2 + 1) * 512);
                c0 = __builtin_amdgcn_wmma_f32_16x16x32_bf16(
                         false, a0, false, b, (short)0, c0, false, false);
                c1 = __builtin_amdgcn_wmma_f32_16x16x32_bf16(
                         false, a1, false, b, (short)0, c1, false, false);
            }
            const float bias = b1[e * HH + np * 16 + n];
            const int klocal = wave * 16 + n;         // k within chunk 0..511
            const int ks2   = klocal >> 5;
            const int kloc2 = klocal & 31;
            const int lidx  = frag_lane(0, kloc2);
            const int slot  = frag_slot(kloc2);
#pragma unroll
            for (int mt = 0; mt < 2; ++mt) {
                const v8f& c = mt ? c1 : c0;
#pragma unroll
                for (int r = 0; r < 8; ++r) {
                    const int mr = mbase + r;
                    const int m  = mt * 16 + mr;
                    float hv = c[r] + bias;
                    hv = hv > 0.f ? hv : 0.f;
                    hv *= wm[m * EE + e];             // fold gate weight into h
                    hA[(ks2 * 2 + mt) * 512 + (lidx + mr) * 16 + slot] = (bf16_t)hv;
                }
            }
        }
        __syncthreads();
        // ---- GEMM2 partial: acc += h_chunk @ W2_chunk ----
#pragma unroll
        for (int j = 0; j < 2; ++j) {
            const int np2 = wave * 2 + j;             // D panel 0..63
            const bf16_t* bp = W2e + ((size_t)np2 * 128 + hc * 16) * 512 + lane * 16;
            const bf16_t* ap = hA + lane * 16;
            for (int kk = 0; kk < 16; ++kk) {
                v16bf b  = *(const v16bf*)(bp + (size_t)kk * 512);
                v16bf a0 = *(const v16bf*)(ap + (kk * 2 + 0) * 512);
                v16bf a1 = *(const v16bf*)(ap + (kk * 2 + 1) * 512);
                acc[j][0] = __builtin_amdgcn_wmma_f32_16x16x32_bf16(
                                false, a0, false, b, (short)0, acc[j][0], false, false);
                acc[j][1] = __builtin_amdgcn_wmma_f32_16x16x32_bf16(
                                false, a1, false, b, (short)0, acc[j][1], false, false);
            }
        }
    }

    // ---- epilogue: atomically accumulate this expert's weighted contribution ----
#pragma unroll
    for (int j = 0; j < 2; ++j) {
        const int dcol = (wave * 2 + j) * 16 + n;
#pragma unroll
        for (int mt = 0; mt < 2; ++mt) {
#pragma unroll
            for (int r = 0; r < 8; ++r) {
                const int m = mt * 16 + mbase + r;
                unsafeAtomicAdd(&out[(size_t)(tok0 + m) * DD + dcol], acc[j][mt][r]);
            }
        }
    }
}

// ---------------- host launch ----------------
extern "C" void kernel_launch(void* const* d_in, const int* in_sizes, int n_in,
                              void* d_out, int out_size, void* d_ws, size_t ws_size,
                              hipStream_t stream) {
    const float* x  = (const float*)d_in[0];
    const float* Wg = (const float*)d_in[1];
    const float* W1 = (const float*)d_in[2];
    const float* b1 = (const float*)d_in[3];
    const float* W2 = (const float*)d_in[4];
    const float* b2 = (const float*)d_in[5];
    float* out = (float*)d_out;

    char* ws = (char*)d_ws;
    const size_t wbytes = (size_t)EE * DD * HH * sizeof(bf16_t);  // 67,108,864
    bf16_t* W1ws  = (bf16_t*)ws;
    bf16_t* W2ws  = (bf16_t*)(ws + wbytes);
    float*  wmask = (float*)(ws + 2 * wbytes);

    const size_t welems = (size_t)EE * DD * HH;                   // 33,554,432
    conv_w1<<<(uint32_t)(welems / 256), 256, 0, stream>>>(W1, W1ws);
    conv_w2<<<(uint32_t)(welems / 256), 256, 0, stream>>>(W2, W2ws);
    gate_kernel<<<NTOK / 2, 64, 0, stream>>>(x, Wg, wmask);
    out_bias_init<<<(NTOK * DD) / 256, 256, 0, stream>>>(b2, wmask, out);
    moe_ffn<<<dim3(NTILE, EE), NTHR, SMEM_BYTES, stream>>>(x, b1, W1ws, W2ws, wmask, out);
}
